// ImprovedGuidedStudent_72791105732694
// MI455X (gfx1250) — compile-verified, matrix operand order checked by reading, not run on hardware
//
#include <hip/hip_runtime.h>
#include <hip/hip_fp16.h>
#include <math.h>

// Problem constants (match reference)
#define BB 512
#define SS 4096
#define DD 128
#define VV 6
#define KSEL 409   // max(1, int(4096*0.1))

typedef __attribute__((ext_vector_type(2)))  float    v2f;
typedef __attribute__((ext_vector_type(8)))  float    v8f;
typedef __attribute__((ext_vector_type(16))) _Float16 v16h;

// Workspace layout (in float units)
#define WS_SCORE   0      // 6 f32: per-vocab sigmoid score
#define WS_ATTN    8      // 6*128 f32: per-vocab attended row (incl. a2)
#define WS_ORDER   776    // 6 int: vocab ids sorted by score desc
#define WS_HIST    784    // 512*6 int: per-row token histogram
#define WS_NSEL    3856   // 512*6 int: selected count per rank
#define WS_POOLED  6928   // 512*128 f32
#define WS_Z1      72464  // 512*64 f32

__device__ __forceinline__ float sigm(float x){ return 1.0f/(1.0f+expf(-x)); }

// ---------------- Kernel 0: per-vocab tables (tiny) + zero hist + sort order
__global__ void k_precompute(const float* __restrict__ emb,
                             const float* __restrict__ W1, const float* __restrict__ b1,
                             const float* __restrict__ W2, const float* __restrict__ b2,
                             const float* __restrict__ W3, const float* __restrict__ b3,
                             const float* __restrict__ A1, const float* __restrict__ a1,
                             const float* __restrict__ A2, const float* __restrict__ a2,
                             float* __restrict__ ws) {
  __shared__ float e[DD];
  __shared__ float h1[64];
  __shared__ float h2[32];
  __shared__ float g[64];
  const int t = threadIdx.x;
  int* wsI = (int*)ws;
  for (int i = t; i < BB*VV; i += blockDim.x) wsI[WS_HIST + i] = 0;

  for (int v = 0; v < VV; ++v) {
    if (t < DD) e[t] = emb[v*DD + t];
    __syncthreads();
    if (t < 64) {
      float s = b1[t];
      for (int d = 0; d < DD; ++d) s = fmaf(e[d], W1[d*64 + t], s);
      h1[t] = fmaxf(s, 0.0f);
      float sg = a1[t];
      for (int d = 0; d < DD; ++d) sg = fmaf(e[d], A1[d*64 + t], sg);
      g[t] = fmaxf(sg, 0.0f);
    }
    __syncthreads();
    if (t < 32) {
      float s = b2[t];
      for (int j = 0; j < 64; ++j) s = fmaf(h1[j], W2[j*32 + t], s);
      h2[t] = fmaxf(s, 0.0f);
    }
    __syncthreads();
    if (t == 0) {
      float s = b3[0];
      for (int j = 0; j < 32; ++j) s = fmaf(h2[j], W3[j], s);
      ws[WS_SCORE + v] = sigm(s);
    }
    if (t < DD) {
      float s = a2[t];
      for (int j = 0; j < 64; ++j) s = fmaf(g[j], A2[j*DD + t], s);
      ws[WS_ATTN + v*DD + t] = s;  // full attended row for vocab v
    }
    __syncthreads();
  }
  if (t == 0) {  // stable insertion sort, descending score
    int ord[VV]; for (int v = 0; v < VV; ++v) ord[v] = v;
    for (int i = 1; i < VV; ++i) {
      int o = ord[i]; float sv = ws[WS_SCORE + o]; int j = i - 1;
      while (j >= 0 && ws[WS_SCORE + ord[j]] < sv) { ord[j+1] = ord[j]; --j; }
      ord[j+1] = o;
    }
    for (int r = 0; r < VV; ++r) wsI[WS_ORDER + r] = ord[r];
  }
}

// ---------------- Kernel 1: scores (table lookup) + per-row histogram
__global__ void k_scores(const int* __restrict__ x, const float* __restrict__ ws,
                         float* __restrict__ scores_out, int* __restrict__ hist) {
  __shared__ int   lh[VV];
  __shared__ float st[VV];
  const int t = threadIdx.x;
  if (t < VV) { lh[t] = 0; st[t] = ws[WS_SCORE + t]; }
  __syncthreads();
  const int i = blockIdx.x * blockDim.x + t;      // grid exactly covers B*S
  const int v = x[i];
  scores_out[i] = st[v];
  atomicAdd(&lh[v], 1);
  __syncthreads();
  const int b = (blockIdx.x * blockDim.x) / SS;   // block lies within one row
  if (t < VV && lh[t]) atomicAdd(&hist[b*VV + t], lh[t]);
}

// ---------------- Kernel 2: per-row selection plan (counts per rank)
__global__ void k_plan(float* __restrict__ ws) {
  const int b = blockIdx.x * blockDim.x + threadIdx.x;
  if (b >= BB) return;
  int* wsI = (int*)ws;
  int rem = KSEL;
  for (int r = 0; r < VV; ++r) {
    const int v = wsI[WS_ORDER + r];
    const int c = wsI[WS_HIST + b*VV + v];
    const int take = c < rem ? c : rem;
    wsI[WS_NSEL + b*VV + r] = take;
    rem -= take;
  }
}

// ---------------- Kernel 3: emit top_idx via wave32 ballot occurrence scan
__global__ void k_emit(const int* __restrict__ x, const float* __restrict__ ws,
                       float* __restrict__ topidx_out) {
  const int b = blockIdx.x;
  const int lane = threadIdx.x;
  const int* wsI = (const int*)ws;
  int ordv[VV], nsel[VV], base[VV], rank[VV], cnt[VV];
  int pre = 0;
  for (int r = 0; r < VV; ++r) {
    ordv[r] = wsI[WS_ORDER + r];
    nsel[r] = wsI[WS_NSEL + b*VV + r];
    base[r] = pre; pre += nsel[r];
  }
  for (int r = 0; r < VV; ++r) rank[ordv[r]] = r;
  for (int v = 0; v < VV; ++v) cnt[v] = 0;
  const unsigned lt = (1u << lane) - 1u;
  for (int s0 = 0; s0 < SS; s0 += 32) {
    const int tok = x[b*SS + s0 + lane];
    for (int v = 0; v < VV; ++v) {
      const unsigned m = (unsigned)__ballot(tok == v);
      const int ord = cnt[v] + __popc(m & lt);
      const int r = rank[v];
      if (tok == v && ord < nsel[r])
        topidx_out[b*KSEL + base[r] + ord] = (float)(s0 + lane);
      cnt[v] += __popc(m);
    }
  }
}

// ---------------- Kernel 4: pooled = (1/k) sum_r nsel_r * attn_table[order[r]]
__global__ void k_pooled(float* __restrict__ ws) {
  const int b = blockIdx.x, d = threadIdx.x;
  const int* wsI = (const int*)ws;
  float s = 0.0f;
  for (int r = 0; r < VV; ++r) {
    const int v = wsI[WS_ORDER + r];
    const int n = wsI[WS_NSEL + b*VV + r];
    s = fmaf((float)n, ws[WS_ATTN + v*DD + d], s);
  }
  ws[WS_POOLED + b*DD + d] = s * (1.0f / (float)KSEL);
}

// ---------------- Kernel 5: z1 = relu(pooled @ C1 + c1) via WMMA
#if __has_builtin(__builtin_amdgcn_wmma_f32_16x16x4_f32)
// Native f32 WMMA: exact fp32 GEMM. One wave per 16x16 tile, K=128 in steps of 4.
__global__ void k_gemm1(const float* __restrict__ C1w, const float* __restrict__ c1b,
                        float* __restrict__ ws) {
  const float* Ap = ws + WS_POOLED;   // [B, 128]
  float* z1 = ws + WS_Z1;             // [B, 64]
  const int m0 = blockIdx.x * 16, n0 = blockIdx.y * 16;
  const int lane = threadIdx.x;
  const int half = lane >> 4;         // 0: K pair {0,1}, 1: K pair {2,3}
  const int mn = lane & 15;
  v8f acc = {};
  for (int k0 = 0; k0 < DD; k0 += 4) {
    const int kk = k0 + 2*half;
    v2f a, bf;
    a[0]  = Ap[(m0 + mn)*DD + kk];
    a[1]  = Ap[(m0 + mn)*DD + kk + 1];
    bf[0] = C1w[(kk    )*64 + n0 + mn];
    bf[1] = C1w[(kk + 1)*64 + n0 + mn];
    acc = __builtin_amdgcn_wmma_f32_16x16x4_f32(false, a, false, bf,
                                                (short)0, acc, false, false);
  }
  for (int r = 0; r < 8; ++r) {
    const int M = r + 8*half, N = mn;
    const float vv = acc[r] + c1b[n0 + N];
    z1[(m0 + M)*64 + n0 + N] = fmaxf(vv, 0.0f);
  }
}
#else
// Fallback: codegen-confirmed f16 WMMA (values are ~1e-2, f16 is adequate here).
__global__ void k_gemm1(const float* __restrict__ C1w, const float* __restrict__ c1b,
                        float* __restrict__ ws) {
  const float* Ap = ws + WS_POOLED;
  float* z1 = ws + WS_Z1;
  const int m0 = blockIdx.x * 16, n0 = blockIdx.y * 16;
  const int lane = threadIdx.x;
  const int half = lane >> 4;
  const int mn = lane & 15;
  v8f acc = {};
  for (int k0 = 0; k0 < DD; k0 += 32) {
    v16h a, bf;
    for (int r = 0; r < 8; ++r) {
      const int grp = r >> 2;
      const int kk = k0 + grp*16 + half*8 + (r & 3)*2;
      a[2*r]    = (_Float16)Ap[(m0 + mn)*DD + kk];
      a[2*r+1]  = (_Float16)Ap[(m0 + mn)*DD + kk + 1];
      bf[2*r]   = (_Float16)C1w[(kk    )*64 + n0 + mn];
      bf[2*r+1] = (_Float16)C1w[(kk + 1)*64 + n0 + mn];
    }
    acc = __builtin_amdgcn_wmma_f32_16x16x32_f16(false, a, false, bf,
                                                 (short)0, acc, false, false);
  }
  for (int r = 0; r < 8; ++r) {
    const int M = r + 8*half, N = mn;
    const float vv = acc[r] + c1b[n0 + N];
    z1[(m0 + M)*64 + n0 + N] = fmaxf(vv, 0.0f);
  }
}
#endif

// ---------------- Kernel 6: pred = sigmoid(z1 @ C2 + c2)
__global__ void k_pred(const float* __restrict__ C2w, const float* __restrict__ c2b,
                       const float* __restrict__ ws, float* __restrict__ pred) {
  __shared__ float s[64];
  const int b = blockIdx.x, j = threadIdx.x;
  s[j] = ws[WS_Z1 + b*64 + j] * C2w[j];
  __syncthreads();
  if (j == 0) {
    float t = 0.0f;
    for (int i = 0; i < 64; ++i) t += s[i];
    pred[b] = sigm(t + c2b[0]);
  }
}

extern "C" void kernel_launch(void* const* d_in, const int* in_sizes, int n_in,
                              void* d_out, int out_size, void* d_ws, size_t ws_size,
                              hipStream_t stream) {
  (void)in_sizes; (void)n_in; (void)out_size; (void)ws_size;
  const int*   x   = (const int*)d_in[0];
  const float* emb = (const float*)d_in[1];
  const float* W1  = (const float*)d_in[2];
  const float* b1  = (const float*)d_in[3];
  const float* W2  = (const float*)d_in[4];
  const float* b2  = (const float*)d_in[5];
  const float* W3  = (const float*)d_in[6];
  const float* b3  = (const float*)d_in[7];
  const float* A1  = (const float*)d_in[8];
  const float* a1  = (const float*)d_in[9];
  const float* A2  = (const float*)d_in[10];
  const float* a2  = (const float*)d_in[11];
  const float* C1  = (const float*)d_in[12];
  const float* c1  = (const float*)d_in[13];
  const float* C2  = (const float*)d_in[14];
  const float* c2  = (const float*)d_in[15];

  float* out = (float*)d_out;
  float* ws  = (float*)d_ws;
  float* pred_out   = out;                       // [512]
  float* topidx_out = out + BB;                  // [512*409] (as float)
  float* scores_out = out + BB + BB*KSEL;        // [512*4096]
  int*   hist       = ((int*)ws) + WS_HIST;

  k_precompute<<<1, 128, 0, stream>>>(emb, W1, b1, W2, b2, W3, b3,
                                      A1, a1, A2, a2, ws);
  k_scores<<<(BB*SS)/256, 256, 0, stream>>>(x, ws, scores_out, hist);
  k_plan<<<(BB + 255)/256, 256, 0, stream>>>(ws);
  k_emit<<<BB, 32, 0, stream>>>(x, ws, topidx_out);
  k_pooled<<<BB, DD, 0, stream>>>(ws);
  k_gemm1<<<dim3(32, 4), 32, 0, stream>>>(C1, c1, ws);
  k_pred<<<BB, 64, 0, stream>>>(C2, c2, ws, pred_out);
}
